// SpatialSelfAttention_28922309771585
// MI455X (gfx1250) — matrix operand order repaired and checked by reading, not compiled, not run on hardware
//
#include <hip/hip_runtime.h>

// ---------------- types matching CDNA5 WMMA builtins ----------------
typedef __attribute__((ext_vector_type(16))) __bf16 v16bf;
typedef __attribute__((ext_vector_type(2)))  __bf16 bf16x2;
typedef __attribute__((ext_vector_type(8)))  float  v8f;
typedef __attribute__((ext_vector_type(8)))  unsigned v8u;

#define DM     128      // d_model
#define NSEQ   512      // N
#define NTOK   16384    // B*P*N
#define NCHUNK 32       // B*P
#define TT     3
#define DKH    32       // per-head dim

// ---------------- bf16 conversion: prefer hardware v_cvt_pk_bf16_f32 ----------------
__device__ __forceinline__ unsigned pack_bf2(float a, float b) {
#if __has_builtin(__builtin_amdgcn_cvt_pk_bf16_f32)
    return __builtin_bit_cast(unsigned, __builtin_amdgcn_cvt_pk_bf16_f32(a, b));
#elif defined(__AMDGCN__)
    bf16x2 r;
    r.x = (__bf16)a;
    r.y = (__bf16)b;
    return __builtin_bit_cast(unsigned, r);
#else
    unsigned ua = __builtin_bit_cast(unsigned, a);
    unsigned ub = __builtin_bit_cast(unsigned, b);
    ua += 0x7FFFu + ((ua >> 16) & 1u);
    ub += 0x7FFFu + ((ub >> 16) & 1u);
    return (ua >> 16) | (ub & 0xFFFF0000u);
#endif
}
__device__ __forceinline__ unsigned short f2bf(float f) {
    return (unsigned short)(pack_bf2(f, 0.0f) & 0xFFFFu);
}
__device__ __forceinline__ v16bf words_to_bf(v8u w) {
    return __builtin_bit_cast(v16bf, w);
}
__device__ __forceinline__ v8f wmma_bf(v16bf a, v16bf b, v8f c) {
    return __builtin_amdgcn_wmma_f32_16x16x32_bf16(false, a, false, b, (short)0, c, false, false);
}

// A-tile (16x32, 16-bit): lane=half*16+r holds row M=r; dword j holds K = (j/4)*16 + half*8 + (j%4)*2, +1
__device__ __forceinline__ v16bf loadA_f32(const float* __restrict__ src, int ld, int r0, int k0) {
    int lane = threadIdx.x & 31, half = lane >> 4, r = lane & 15;
    v8u w;
    #pragma unroll
    for (int j = 0; j < 8; ++j) {
        int kc = ((j >> 2) << 4) + (half << 3) + ((j & 3) << 1);
        const float* p = src + (size_t)(r0 + r) * ld + k0 + kc;
        w[j] = pack_bf2(p[0], p[1]);
    }
    return words_to_bf(w);
}
__device__ __forceinline__ v16bf loadA_bf16(const unsigned short* __restrict__ src, int ld, int r0, int k0) {
    int lane = threadIdx.x & 31, half = lane >> 4, r = lane & 15;
    v8u w;
    #pragma unroll
    for (int j = 0; j < 8; ++j) {
        int kc = ((j >> 2) << 4) + (half << 3) + ((j & 3) << 1);
        w[j] = *(const unsigned*)(src + (size_t)(r0 + r) * ld + k0 + kc);
    }
    return words_to_bf(w);
}
// B-tile (32x16, 16-bit) with B[k][n] = src[n0+n][k0+k] (pairs contiguous along k):
// lane: n = lane%16, dword j holds K = (j/4)*16 + half*8 + (j%4)*2, +1
__device__ __forceinline__ v16bf loadBT_f32(const float* __restrict__ src, int ld, int n0, int k0) {
    int lane = threadIdx.x & 31, half = lane >> 4, n = lane & 15;
    v8u w;
    #pragma unroll
    for (int j = 0; j < 8; ++j) {
        int kc = ((j >> 2) << 4) + (half << 3) + ((j & 3) << 1);
        const float* p = src + (size_t)(n0 + n) * ld + k0 + kc;
        w[j] = pack_bf2(p[0], p[1]);
    }
    return words_to_bf(w);
}
__device__ __forceinline__ v16bf loadBT_bf16(const unsigned short* __restrict__ src, int ld, int n0, int k0) {
    int lane = threadIdx.x & 31, half = lane >> 4, n = lane & 15;
    v8u w;
    #pragma unroll
    for (int j = 0; j < 8; ++j) {
        int kc = ((j >> 2) << 4) + (half << 3) + ((j & 3) << 1);
        w[j] = *(const unsigned*)(src + (size_t)(n0 + n) * ld + k0 + kc);
    }
    return words_to_bf(w);
}
// 16-lane (half-wave) reductions — wave32, width 16
__device__ __forceinline__ float redmax16(float v) {
    #pragma unroll
    for (int i = 1; i < 16; i <<= 1) v = fmaxf(v, __shfl_xor(v, i, 16));
    return v;
}
__device__ __forceinline__ float redsum16(float v) {
    #pragma unroll
    for (int i = 1; i < 16; i <<= 1) v += __shfl_xor(v, i, 16);
    return v;
}

// =====================================================================
// Kernel A: QKV projections. 9 GEMMs [16384x128]x[128->128].
// grid: 9 mats * 128 row-groups; block=256 (8 waves); wave = 16 rows x 128 cols
// Q,K -> [t][tok][128] bf16 ; V -> transposed [t][chunk][d=128][n=512] bf16
// =====================================================================
__global__ void __launch_bounds__(256) proj_kernel(
    const float* __restrict__ X, const float* __restrict__ Wq,
    const float* __restrict__ Wk, const float* __restrict__ Wv,
    unsigned short* __restrict__ Qb, unsigned short* __restrict__ Kb,
    unsigned short* __restrict__ Vt)
{
    int wave = threadIdx.x >> 5, lane = threadIdx.x & 31;
    int half = lane >> 4, nl = lane & 15;
    int mat = blockIdx.x >> 7;                  // 0..8
    int t = mat / 3, which = mat % 3;           // 0=Q 1=K 2=V
    const float* W = (which == 0 ? Wq : which == 1 ? Wk : Wv) + (size_t)t * DM * DM;
    int r0 = ((int)(blockIdx.x & 127) * 8 + wave) * 16;

    v8f acc[8];
    #pragma unroll
    for (int i = 0; i < 8; ++i)
        #pragma unroll
        for (int e = 0; e < 8; ++e) acc[i][e] = 0.0f;

    #pragma unroll
    for (int kb = 0; kb < 4; ++kb) {
        v16bf a = loadA_f32(X, DM, r0, kb * 32);
        #pragma unroll
        for (int ot = 0; ot < 8; ++ot) {
            v16bf b = loadBT_f32(W, DM, ot * 16, kb * 32);  // B[k][n]=W[n][k]
            acc[ot] = wmma_bf(a, b, acc[ot]);
        }
    }

    if (which < 2) {
        unsigned short* dst = (which == 0 ? Qb : Kb) + (size_t)t * NTOK * DM;
        #pragma unroll
        for (int ot = 0; ot < 8; ++ot)
            #pragma unroll
            for (int j = 0; j < 8; ++j) {
                int tok = r0 + half * 8 + j;
                dst[(size_t)tok * DM + ot * 16 + nl] = f2bf(acc[ot][j]);
            }
    } else {
        int chunk = r0 / NSEQ, nloc0 = r0 % NSEQ;
        unsigned short* dst = Vt + ((size_t)t * NCHUNK + chunk) * DM * NSEQ;
        #pragma unroll
        for (int ot = 0; ot < 8; ++ot)
            #pragma unroll
            for (int j = 0; j < 8; ++j) {
                int n = nloc0 + half * 8 + j;
                dst[(size_t)(ot * 16 + nl) * NSEQ + n] = f2bf(acc[ot][j]);
            }
    }
}

// =====================================================================
// Kernel B: fused masked attention with post-softmax tm reweighting.
// grid: 384 (t,bp,h) combos * 4 row-slices; block=256; wave = 16 query rows
// out: AttO [tok][T*128] bf16 (f = t*128 + h*32 + dk)
// =====================================================================
__global__ void __launch_bounds__(256) attn_kernel(
    const unsigned short* __restrict__ Qb, const unsigned short* __restrict__ Kb,
    const unsigned short* __restrict__ Vt, const float* __restrict__ TMall,
    unsigned short* __restrict__ AttO)
{
    __shared__ __align__(16) unsigned short Pst[8][16][32]; // per-wave C->A transpose staging
    int wave = threadIdx.x >> 5, lane = threadIdx.x & 31;
    int half = lane >> 4, nl = lane & 15;

    int rowblk = blockIdx.x & 3;
    int combo  = blockIdx.x >> 2;               // 0..383
    int h  = combo & 3;
    int bp = (combo >> 2) & 31;
    int t  = combo >> 7;                        // 0..2
    int tm_idx = (t == 0) ? 0 : (t == 1) ? 1 : 3;   // WHICH_IDX = [0,1,3]

    const float* tm = TMall + (size_t)tm_idx * NSEQ * NSEQ;
    const unsigned short* Q = Qb + ((size_t)t * NTOK + (size_t)bp * NSEQ) * DM;
    const unsigned short* K = Kb + ((size_t)t * NTOK + (size_t)bp * NSEQ) * DM;
    const unsigned short* V = Vt + ((size_t)t * NCHUNK + bp) * (size_t)DM * NSEQ;

    int n0 = rowblk * 128 + wave * 16;
    const float scale = 0.17677669529663687f;   // 1/sqrt(32)
    const float NEG_INF = -__builtin_inff();

    v16bf qa = loadA_bf16(Q, DM, n0, h * DKH);  // A = Q[16 x 32]

    float mrow[8], lrow[8];
    v8f o0, o1;
    #pragma unroll
    for (int j = 0; j < 8; ++j) { mrow[j] = NEG_INF; lrow[j] = 0.f; o0[j] = 0.f; o1[j] = 0.f; }

    for (int mb = 0; mb < NSEQ; mb += 32) {
        v16bf kb0 = loadBT_bf16(K, DM, mb,      h * DKH);   // B[dk][m] = K[m][dk]
        v16bf kb1 = loadBT_bf16(K, DM, mb + 16, h * DKH);
        v8f z;
        #pragma unroll
        for (int e = 0; e < 8; ++e) z[e] = 0.f;
        v8f s0 = wmma_bf(qa, kb0, z);
        v8f s1 = wmma_bf(qa, kb1, z);

        float t0[8], t1[8];
        #pragma unroll
        for (int j = 0; j < 8; ++j) {
            int n = n0 + half * 8 + j;
            t0[j] = tm[(size_t)n * NSEQ + mb + nl];
            t1[j] = tm[(size_t)n * NSEQ + mb + 16 + nl];
            s0[j] = (t0[j] != 0.f) ? s0[j] * scale : NEG_INF;
            s1[j] = (t1[j] != 0.f) ? s1[j] * scale : NEG_INF;
        }

        #pragma unroll
        for (int j = 0; j < 8; ++j) {
            float tmax = redmax16(fmaxf(s0[j], s1[j]));
            float mnew = fmaxf(mrow[j], tmax);
            bool dead = (mnew == NEG_INF);
            float corr = dead ? 1.0f : __expf(mrow[j] - mnew);
            float p0 = dead ? 0.0f : __expf(s0[j] - mnew);
            float p1 = dead ? 0.0f : __expf(s1[j] - mnew);
            lrow[j] = lrow[j] * corr + redsum16(p0 + p1);   // denom: plain softmax
            mrow[j] = mnew;
            o0[j] *= corr; o1[j] *= corr;
            int row = half * 8 + j;                          // stage p*tm for P x V
            unsigned pk = pack_bf2(p0 * t0[j], p1 * t1[j]);
            Pst[wave][row][nl]      = (unsigned short)(pk & 0xFFFFu);
            Pst[wave][row][16 + nl] = (unsigned short)(pk >> 16);
        }

        v16bf pa;                                            // read back in A layout
        {
            v8u w;
            #pragma unroll
            for (int j = 0; j < 8; ++j) {
                int kc = ((j >> 2) << 4) + (half << 3) + ((j & 3) << 1);
                w[j] = *(const unsigned*)&Pst[wave][nl][kc];
            }
            pa = words_to_bf(w);
        }

        v16bf vb0 = loadBT_bf16(V, NSEQ, h * DKH + 0,  mb);  // B[m][dk] = Vt[dk][m]
        v16bf vb1 = loadBT_bf16(V, NSEQ, h * DKH + 16, mb);
        o0 = wmma_bf(pa, vb0, o0);
        o1 = wmma_bf(pa, vb1, o1);
    }

    #pragma unroll
    for (int j = 0; j < 8; ++j) {
        float inv = 1.0f / lrow[j];             // diag of tm is nonzero -> lrow > 0
        int tok = bp * NSEQ + n0 + half * 8 + j;
        size_t base = (size_t)tok * (TT * DM) + (size_t)t * DM + h * DKH;
        unsigned pk = pack_bf2(o0[j] * inv, o1[j] * inv);
        AttO[base + nl]      = (unsigned short)(pk & 0xFFFFu);
        AttO[base + 16 + nl] = (unsigned short)(pk >> 16);
    }
}

// =====================================================================
// Kernel C: out = LN( AttO[16384x384] @ Wo^T + X )
// grid: 128; block=256; wave = 16 rows x 128 cols; 12 k-steps of 32
// =====================================================================
__global__ void __launch_bounds__(256) outproj_ln_kernel(
    const unsigned short* __restrict__ AttO, const float* __restrict__ Wo,
    const float* __restrict__ X, const float* __restrict__ gamma,
    const float* __restrict__ beta, float* __restrict__ Out)
{
    int wave = threadIdx.x >> 5, lane = threadIdx.x & 31;
    int half = lane >> 4, nl = lane & 15;
    int r0 = ((int)blockIdx.x * 8 + wave) * 16;

    v8f acc[8];
    #pragma unroll
    for (int i = 0; i < 8; ++i)
        #pragma unroll
        for (int e = 0; e < 8; ++e) acc[i][e] = 0.0f;

    #pragma unroll
    for (int kb = 0; kb < 12; ++kb) {
        v16bf a = loadA_bf16(AttO, TT * DM, r0, kb * 32);
        #pragma unroll
        for (int ot = 0; ot < 8; ++ot) {
            v16bf b = loadBT_f32(Wo, TT * DM, ot * 16, kb * 32); // B[f][d]=Wo[d][f]
            acc[ot] = wmma_bf(a, b, acc[ot]);
        }
    }

    // residual add
    #pragma unroll
    for (int ot = 0; ot < 8; ++ot)
        #pragma unroll
        for (int j = 0; j < 8; ++j) {
            int tok = r0 + half * 8 + j;
            acc[ot][j] += X[(size_t)tok * DM + ot * 16 + nl];
        }

    // layernorm per row (128 values: 8 tiles in-lane x 16 lanes)
    #pragma unroll
    for (int j = 0; j < 8; ++j) {
        float s1 = 0.f, s2 = 0.f;
        #pragma unroll
        for (int ot = 0; ot < 8; ++ot) { float v = acc[ot][j]; s1 += v; s2 += v * v; }
        s1 = redsum16(s1);
        s2 = redsum16(s2);
        float mu  = s1 * (1.0f / DM);
        float var = fmaxf(s2 * (1.0f / DM) - mu * mu, 0.0f);
        float rs  = rsqrtf(var + 1e-5f);
        int tok = r0 + half * 8 + j;
        #pragma unroll
        for (int ot = 0; ot < 8; ++ot) {
            int col = ot * 16 + nl;
            Out[(size_t)tok * DM + col] = (acc[ot][j] - mu) * rs * gamma[col] + beta[col];
        }
    }
}

// =====================================================================
extern "C" void kernel_launch(void* const* d_in, const int* in_sizes, int n_in,
                              void* d_out, int out_size, void* d_ws, size_t ws_size,
                              hipStream_t stream) {
    const float* X  = (const float*)d_in[0];
    // d_in[1] (c_inputs) unused by the reference
    const float* TM = (const float*)d_in[2];
    const float* Wq = (const float*)d_in[3];
    const float* Wk = (const float*)d_in[4];
    const float* Wv = (const float*)d_in[5];
    const float* Wo = (const float*)d_in[6];
    const float* g  = (const float*)d_in[7];
    const float* be = (const float*)d_in[8];
    float* out = (float*)d_out;

    const size_t QKV_ELEMS = (size_t)TT * NTOK * DM;          // bf16 elems
    unsigned short* Qb   = (unsigned short*)d_ws;
    unsigned short* Kb   = Qb + QKV_ELEMS;
    unsigned short* Vt   = Kb + QKV_ELEMS;
    unsigned short* AttO = Vt + QKV_ELEMS;                    // [NTOK][TT*DM]

    proj_kernel<<<dim3(9 * 128), dim3(256), 0, stream>>>(X, Wq, Wk, Wv, Qb, Kb, Vt);
    attn_kernel<<<dim3(384 * 4), dim3(256), 0, stream>>>(Qb, Kb, Vt, TM, AttO);
    outproj_ln_kernel<<<dim3(128), dim3(256), 0, stream>>>(AttO, Wo, X, g, be, out);
}
